// BCellIRTActor_18021682774618
// MI455X (gfx1250) — matrix operand order repaired and checked by reading, not compiled, not run on hardware
//
#include <hip/hip_runtime.h>
#include <hip/hip_bf16.h>

typedef __attribute__((ext_vector_type(16))) __bf16 v16bf;
typedef __attribute__((ext_vector_type(8)))  __bf16 v8bf;
typedef __attribute__((ext_vector_type(8)))  float  v8f;

// ---------------------------------------------------------------------------
// Kernel 1: pack weights (fp32 row-major) into bf16 WMMA B-fragment order.
// Order used here (contiguous k, same as A): lane l<16 -> N=l,    K = base + e        (e=0..15)
//                           lane l>=16 -> N=l-16, K = base + 16 + e
// p1  : we1 (274->288 K-pad) x 256           : 9 kt x 16 nt = 144 frags
// p2e : [we2 (256x768) | W_sim (256x48) | 0] : 8 kt x 52 nt = 416 frags
//       W_sim[k][i*8+j] = sum_d we2[k][i*128+d] * Kn[j][d]  (Kn normalized)
//       nt==51 is a zero pad tile so 2 rowtiles x 52 nt = 104 = 8 waves x 13.
// ---------------------------------------------------------------------------
__global__ __launch_bounds__(256)
void pack_weights(const float* __restrict__ we1, const float* __restrict__ we2,
                  const float* __restrict__ pk,
                  __bf16* __restrict__ p1, __bf16* __restrict__ p2e)
{
    const int fid  = blockIdx.x * 8 + (threadIdx.x >> 5);
    const int lane = threadIdx.x & 31;
    const int koff = ((lane >> 4) & 1) * 16;
    if (fid < 144) {
        const int kt = fid / 16, nt = fid % 16;
        const int n  = nt * 16 + (lane & 15);
        __bf16* dst = p1 + (size_t)fid * 512 + lane * 16;
        const int kb = kt * 32 + koff;
#pragma unroll
        for (int e = 0; e < 16; e++) {
            const int k = kb + e;
            dst[e] = (__bf16)((k < 274) ? we1[(size_t)k * 256 + n] : 0.0f);
        }
    } else if (fid < 144 + 416) {
        const int f2 = fid - 144;
        const int kt = f2 / 52, nt = f2 % 52;
        const int kb = kt * 32 + koff;
        __bf16* dst = p2e + (size_t)f2 * 512 + lane * 16;
        if (nt < 48) {
            const int n = nt * 16 + (lane & 15);
#pragma unroll
            for (int e = 0; e < 16; e++)
                dst[e] = (__bf16)we2[(size_t)(kb + e) * 768 + n];
        } else if (nt < 51) {
            const int c = (nt - 48) * 16 + (lane & 15);     // 0..47
            const int i = c >> 3, j = c & 7;
            float s = 0.0f;
            for (int d = 0; d < 128; d++) { const float x = pk[j * 128 + d]; s += x * x; }
            const float inv = 1.0f / (sqrtf(s) + 1e-8f);
            for (int e = 0; e < 16; e++) {
                const int k = kb + e;
                float acc = 0.0f;
                for (int d = 0; d < 128; d++)
                    acc += we2[(size_t)k * 768 + i * 128 + d] * pk[j * 128 + d];
                dst[e] = (__bf16)(acc * inv);
            }
        } else {                                            // zero pad tile
#pragma unroll
            for (int e = 0; e < 16; e++) dst[e] = (__bf16)0.0f;
        }
    }
}

// ---------------------------------------------------------------------------
// Kernel 2: batch-independent conc table (8x30) and sim bias (be2 . Kn, 48).
// ---------------------------------------------------------------------------
__global__ __launch_bounds__(1024)
void conc_kernel(const float* __restrict__ pk,  const float* __restrict__ wd1,
                 const float* __restrict__ bd1, const float* __restrict__ wd2,
                 const float* __restrict__ bd2, const float* __restrict__ be2,
                 float* __restrict__ conc, float* __restrict__ simb)
{
    __shared__ float sHd[8 * 128];
    const int t = threadIdx.x;
    {
        const int j = t >> 7, i = t & 127;
        float acc = bd1[j * 128 + i];
        for (int d = 0; d < 128; d++)
            acc += pk[j * 128 + d] * wd1[((size_t)(j * 128 + d)) * 128 + i];
        sHd[t] = fmaxf(acc, 0.0f);
    }
    __syncthreads();
    if (t < 240) {
        const int j = t / 30, a = t % 30;
        float c = bd2[j * 30 + a];
        for (int h = 0; h < 128; h++)
            c += sHd[j * 128 + h] * wd2[((size_t)(j * 128 + h)) * 30 + a];
        conc[j * 30 + a] = (c > 20.0f) ? c : log1pf(expf(c));   // softplus
    } else if (t < 288) {
        const int c = t - 240;                  // 0..47
        const int i = c >> 3, j = c & 7;
        float s = 0.0f;
        for (int d = 0; d < 128; d++) { const float x = pk[j * 128 + d]; s += x * x; }
        const float inv = 1.0f / (sqrtf(s) + 1e-8f);
        float acc = 0.0f;
        for (int d = 0; d < 128; d++) acc += be2[i * 128 + d] * pk[j * 128 + d];
        simb[c] = acc * inv;
    }
}

// ---------------------------------------------------------------------------
// Kernel 3: fused actor forward. 32 batch rows / block, 256 threads = 8 waves.
// Wave wv owns rowtile rt = wv>>2 and N-tiles nt = (wv&3) + 4*s.
// ---------------------------------------------------------------------------
__global__ __launch_bounds__(256)
void actor_main(const float* __restrict__ state, const float* __restrict__ fitness,
                const float* __restrict__ be1,   const float* __restrict__ ln_g,
                const float* __restrict__ ln_b,  const float* __restrict__ be2,
                const float* __restrict__ proto_keys,
                const float* __restrict__ wt,    const float* __restrict__ bt,
                const float* __restrict__ wc,    const float* __restrict__ bc,
                const float* __restrict__ w_prev,const float* __restrict__ crisis_bias,
                const __bf16* __restrict__ pw1,  const __bf16* __restrict__ pw2e,
                const float* __restrict__ conc,  const float* __restrict__ simb,
                float* __restrict__ out)
{
    // Big buffer aliased over time:
    //  [t0] sAbf  : 32x288 bf16 state tile (18432 B, in first 32768 B) -> dead after GEMM1
    //  [t1] sHpre : 32x256 f32 pre-LN (32768 B)
    //  [t1] sHbf  : 32x256 bf16 post-LN at +32768 (16384 B)
    __shared__ float sBig[32 * 256 + 32 * 128];      // 49152 B
    __shared__ float sHt[32 * 128];
    __shared__ float sKn[8 * 128];
    __shared__ float sSim[32 * 48];
    __shared__ float sNormSq[32 * 6];
    __shared__ float sLogK[32 * 48];
    __shared__ float sU[32 * 6];
    __shared__ float sV[32 * 8];
    __shared__ float sWot[32 * 8];
    __shared__ float sMf[32 * 12];
    __shared__ float sCrisis[32], sMu[32], sRstd[32], sDnInv[32];
    __shared__ float sDsim[32 * 8];
    __shared__ float sRedS[256], sRedQ[256];
    __shared__ float sConc[240];
    __shared__ float sSimB[48];
    __shared__ float sKnorm[8];

    __bf16* sAbf  = (__bf16*)sBig;
    float*  sHpre = sBig;
    __bf16* sHbf  = (__bf16*)(sBig + 8192);

    const int t    = threadIdx.x;
    const int lane = t & 31;
    const int wv   = t >> 5;
    const int ln15 = lane & 15;
    const int hi   = (lane >> 4) & 1;
    const int rt   = wv >> 2;            // fixed rowtile per wave
    const int nq   = wv & 3;             // N-tile stride phase per wave
    const int r0   = blockIdx.x * 32;

    // ---- Phase 0: loads, zeroing, per-row features -------------------------
    for (int i = t; i < 32 * 288; i += 256) {
        const int row = i / 288, col = i - row * 288;
        const float v = (col < 274) ? state[(size_t)(r0 + row) * 274 + col] : 0.0f;
        sAbf[i] = (__bf16)v;
    }
    if (t < 240) sConc[t] = conc[t];
    if (t < 48)  sSimB[t] = simb[t];
    if (t < 192) sNormSq[t] = 0.0f;
    if (t >= 32 && t < 40) {
        const int j = t - 32;
        float s = 0.0f;
        for (int d = 0; d < 128; d++) { const float x = proto_keys[j * 128 + d]; s += x * x; }
        sKnorm[j] = 1.0f / (sqrtf(s) + 1e-8f);
    }
    if (t < 32) {
        const float* g = state + (size_t)(r0 + t) * 274;
        const float bal = g[0];
        float ps = 0.0f, pq = 0.0f, tv = 0.0f;
        for (int k = 0; k < 30; k++) {
            const float p = g[1 + k];
            ps += p; pq += p * p; tv += p * g[31 + k];
        }
        const float mean = ps * (1.0f / 30.0f);
        const float var  = (pq - ps * ps * (1.0f / 30.0f)) * (1.0f / 29.0f);  // ddof=1
        const float sd   = sqrtf(fmaxf(var, 0.0f)) + 1e-8f;
        tv += bal;
        sMf[t * 12 + 0] = bal; sMf[t * 12 + 1] = mean;
        sMf[t * 12 + 2] = sd;  sMf[t * 12 + 3] = bal / (tv + 1e-8f);
        const int TI[8] = {61, 91, 121, 151, 181, 211, 241, 271};
#pragma unroll
        for (int k = 0; k < 8; k++) sMf[t * 12 + 4 + k] = g[TI[k]];
        const float ds = g[272], cv = g[273];
        const float dss = 1.0f / (1.0f + expf(-ds * 10.0f));
        const float cvs = 1.0f / (1.0f + expf(-fabsf(cv) * 50.0f));
        sCrisis[t] = 0.25f * dss + 0.15f * cvs + crisis_bias[0];   // + 0.6*cb below
    }
    __syncthreads();

    // ---- Phase 1: h_t = relu(mf@wt+bt);  Kn = normalized protos ------------
    {
        const int col = t & 127;
        const int rb  = (t >> 7) * 16;
        for (int row = rb; row < rb + 16; row++) {
            float acc = bt[col];
#pragma unroll
            for (int k = 0; k < 12; k++) acc += sMf[row * 12 + k] * wt[k * 128 + col];
            sHt[row * 128 + col] = fmaxf(acc, 0.0f);
        }
        for (int i = t; i < 8 * 128; i += 256) sKn[i] = proto_keys[i] * sKnorm[i >> 7];
    }
    __syncthreads();

    // ---- Phase 2: crisis level; danger inv-norm ----------------------------
    if (t < 32) {
        float s = bc[0], q = 0.0f;
        for (int k = 0; k < 128; k++) {
            const float h = sHt[t * 128 + k];
            s += h * wc[k]; q += h * h;
        }
        sCrisis[t] += 0.6f * (1.0f / (1.0f + expf(-s)));
        sDnInv[t] = 1.0f / (sqrtf(q) + 1e-8f);
    }
    __syncthreads();

    // ---- Phase 3: dsim[row][j] ---------------------------------------------
    {
        const int row = t >> 3, j = t & 7;
        float s = 0.0f;
        for (int k = 0; k < 128; k++) s += sHt[row * 128 + k] * sKn[j * 128 + k];
        sDsim[row * 8 + j] = s * sDnInv[row];
    }
    __syncthreads();

    // ---- GEMM1: 32x288(bf16) @ we1 + be1 -----------------------------------
    // wave wv: rowtile rt, N-tiles nt = nq + 4*s (s=0..3) -> all 16 tiles.
    v8f acc1[4] = {};
    for (int kt = 0; kt < 9; kt++) {
        const __bf16* ap = sAbf + (rt * 16 + ln15) * 288 + kt * 32 + hi * 16;
        const v8bf lo = *(const v8bf*)ap;
        const v8bf hh = *(const v8bf*)(ap + 8);
        v16bf a;
#pragma unroll
        for (int e = 0; e < 8; e++) { a[e] = lo[e]; a[e + 8] = hh[e]; }
#pragma unroll
        for (int s = 0; s < 4; s++) {
            const int nt = nq + 4 * s;
            const v16bf b = *(const v16bf*)(pw1 + ((size_t)(kt * 16 + nt) * 32 + lane) * 16);
            acc1[s] = __builtin_amdgcn_wmma_f32_16x16x32_bf16(
                false, a, false, b, (short)0, acc1[s], false, false);
        }
    }
    __syncthreads();                               // sAbf dead -> reuse as sHpre
#pragma unroll
    for (int s = 0; s < 4; s++) {
        const int nt  = nq + 4 * s;
        const int col = nt * 16 + ln15;
        const float b1 = be1[col];
#pragma unroll
        for (int r = 0; r < 8; r++)
            sHpre[(rt * 16 + r + hi * 8) * 256 + col] = acc1[s][r] + b1;
    }
    __syncthreads();

    // ---- LayerNorm + ReLU -> bf16 ------------------------------------------
    {
        const int row = t >> 3, seg = t & 7;
        float s = 0.0f, q = 0.0f;
        for (int c = seg * 32; c < seg * 32 + 32; c++) {
            const float x = sHpre[row * 256 + c];
            s += x; q += x * x;
        }
        sRedS[t] = s; sRedQ[t] = q;
    }
    __syncthreads();
    if (t < 32) {
        float s = 0.0f, q = 0.0f;
#pragma unroll
        for (int k = 0; k < 8; k++) { s += sRedS[t * 8 + k]; q += sRedQ[t * 8 + k]; }
        const float mu  = s * (1.0f / 256.0f);
        const float var = q * (1.0f / 256.0f) - mu * mu;
        sMu[t]   = mu;
        sRstd[t] = rsqrtf(fmaxf(var, 0.0f) + 1e-5f);
    }
    __syncthreads();
    for (int i = t; i < 32 * 256; i += 256) {
        const int row = i >> 8, col = i & 255;
        const float x = (sHpre[i] - sMu[row]) * sRstd[row] * ln_g[col] + ln_b[col];
        sHbf[i] = (__bf16)fmaxf(x, 0.0f);
    }
    __syncthreads();

    // ---- GEMM2: h(32x256) @ [we2 | W_sim | pad] ----------------------------
    // wave wv: rowtile rt, N-tiles nt = nq + 4*s (s=0..12) -> all 52 tiles.
    // nt<48: E columns (used only for chunk-norms); 48..50: sim; 51: pad.
    v8f acc2[13] = {};
    for (int kt = 0; kt < 8; kt++) {
        const __bf16* ap = sHbf + (rt * 16 + ln15) * 256 + kt * 32 + hi * 16;
        const v8bf lo = *(const v8bf*)ap;
        const v8bf hh = *(const v8bf*)(ap + 8);
        v16bf a;
#pragma unroll
        for (int e = 0; e < 8; e++) { a[e] = lo[e]; a[e + 8] = hh[e]; }
#pragma unroll
        for (int s = 0; s < 13; s++) {
            const int nt = nq + 4 * s;
            const v16bf b = *(const v16bf*)(pw2e + ((size_t)(kt * 52 + nt) * 32 + lane) * 16);
            acc2[s] = __builtin_amdgcn_wmma_f32_16x16x32_bf16(
                false, a, false, b, (short)0, acc2[s], false, false);
        }
    }
    // chunk-norm reduction (shfl butterfly over 16-lane N group) + sim store
#pragma unroll
    for (int s = 0; s < 13; s++) {
        const int nt = nq + 4 * s;
        if (nt < 48) {
            const int col   = nt * 16 + ln15;
            const int chunk = nt >> 3;
            const float b2  = be2[col];
#pragma unroll
            for (int r = 0; r < 8; r++) {
                const float x = acc2[s][r] + b2;
                float sq = x * x;
                sq += __shfl_xor(sq, 1, 32);
                sq += __shfl_xor(sq, 2, 32);
                sq += __shfl_xor(sq, 4, 32);
                sq += __shfl_xor(sq, 8, 32);
                if (ln15 == 0)
                    atomicAdd(&sNormSq[(rt * 16 + r + hi * 8) * 6 + chunk], sq);
            }
        } else if (nt < 51) {
            const int c = (nt - 48) * 16 + ln15;         // 0..47 = i*8+j
            const float sb = sSimB[c];
#pragma unroll
            for (int r = 0; r < 8; r++)
                sSim[(rt * 16 + r + hi * 8) * 48 + c] = acc2[s][r] + sb;
        }
        // nt == 51: pad tile, discard
    }
    __syncthreads();

    // ---- logK = 20*(sim + 0.1*dsim - 1);  init u, v ------------------------
    {
        const int row = t >> 3, j = t & 7;
        const float dsh = 0.1f * sDsim[row * 8 + j];
#pragma unroll
        for (int i = 0; i < 6; i++) {
            const float inv = 1.0f / (sqrtf(sNormSq[row * 6 + i]) + 1e-8f);
            const float sim = sSim[row * 48 + i * 8 + j] * inv;
            sLogK[row * 48 + i * 8 + j] = 20.0f * (sim + dsh - 1.0f);
        }
        sV[row * 8 + j] = 0.0f;
        if (t < 192) sU[t] = 0.0f;
    }
    __syncthreads();

    // ---- Sinkhorn (10 iters, stabilized) -----------------------------------
    const float log_mu = -logf(6.0f);
    const float log_nu = -logf(8.0f);
    for (int it = 0; it < 10; it++) {
        if (t < 192) {
            const int row = t / 6, i = t % 6;
            const float* lk = sLogK + row * 48 + i * 8;
            float mx = -1e30f;
#pragma unroll
            for (int j = 0; j < 8; j++) mx = fmaxf(mx, lk[j] + sV[row * 8 + j]);
            float s = 0.0f;
#pragma unroll
            for (int j = 0; j < 8; j++) s += expf(lk[j] + sV[row * 8 + j] - mx);
            sU[row * 6 + i] = log_mu - (mx + logf(s));
        }
        __syncthreads();
        {
            const int row = t >> 3, j = t & 7;
            float mx = -1e30f;
#pragma unroll
            for (int i = 0; i < 6; i++)
                mx = fmaxf(mx, sLogK[row * 48 + i * 8 + j] + sU[row * 6 + i]);
            float s = 0.0f;
#pragma unroll
            for (int i = 0; i < 6; i++)
                s += expf(sLogK[row * 48 + i * 8 + j] + sU[row * 6 + i] - mx);
            sV[row * 8 + j] = log_nu - (mx + logf(s));
        }
        __syncthreads();
    }
    {
        const int row = t >> 3, j = t & 7;
        float s = 0.0f;
#pragma unroll
        for (int i = 0; i < 6; i++)
            s += expf(sLogK[row * 48 + i * 8 + j] + sU[row * 6 + i] + sV[row * 8 + j]);
        sWot[row * 8 + j] = s;
    }
    __syncthreads();

    // ---- replicator mix + Dirichlet head + softmax -> out ------------------
    if (t < 32) {
        const int row = t;
        float wo[8], sum = 0.0f;
#pragma unroll
        for (int j = 0; j < 8; j++) { wo[j] = sWot[row * 8 + j]; sum += wo[j]; }
        const float invo = 1.0f / (sum + 1e-8f);
        const float crisis = sCrisis[row];
        const float eta = 0.05f + 0.15f * crisis;
        const float* fit = fitness + (size_t)(r0 + row) * 8;
        float wr[8], sr = 0.0f;
#pragma unroll
        for (int j = 0; j < 8; j++) { wr[j] = w_prev[j] * expf(eta * fit[j]); sr += wr[j]; }
        const float invr = 1.0f / (sr + 1e-8f);
        const float alpha = fminf(fmaxf(0.3f - 0.24f * crisis, 0.06f), 0.3f);
        float w[8], sw = 0.0f;
#pragma unroll
        for (int j = 0; j < 8; j++) {
            w[j] = (1.0f - alpha) * wr[j] * invr + alpha * wo[j] * invo;
            sw += w[j];
        }
        const float invw = 1.0f / (sw + 1e-8f);
        float mixed[30], mx = -1e30f;
        for (int a = 0; a < 30; a++) {
            float c = 1.0f;
#pragma unroll
            for (int j = 0; j < 8; j++) c += w[j] * invw * sConc[j * 30 + a];
            mixed[a] = c; mx = fmaxf(mx, c);
        }
        float se = 0.0f;
        for (int a = 0; a < 30; a++) { mixed[a] = expf(mixed[a] - mx); se += mixed[a]; }
        const float ise = 1.0f / se;
        float s2 = 0.0f;
        for (int a = 0; a < 30; a++) {
            const float p = fminf(fmaxf(mixed[a] * ise, 0.0f), 1.0f);
            mixed[a] = p; s2 += p;
        }
        const float is2 = 1.0f / (s2 + 1e-8f);
        float* o = out + (size_t)(r0 + row) * 30;
        for (int a = 0; a < 30; a++) o[a] = mixed[a] * is2;
    }
}

// ---------------------------------------------------------------------------
extern "C" void kernel_launch(void* const* d_in, const int* in_sizes, int n_in,
                              void* d_out, int out_size, void* d_ws, size_t ws_size,
                              hipStream_t stream)
{
    (void)in_sizes; (void)n_in; (void)out_size; (void)ws_size;
    const float* state  = (const float*)d_in[0];
    const float* fitness= (const float*)d_in[1];
    const float* we1    = (const float*)d_in[2];
    const float* be1    = (const float*)d_in[3];
    const float* ln_g   = (const float*)d_in[4];
    const float* ln_b   = (const float*)d_in[5];
    const float* we2    = (const float*)d_in[6];
    const float* be2    = (const float*)d_in[7];
    const float* pk     = (const float*)d_in[8];
    const float* wd1    = (const float*)d_in[9];
    const float* bd1    = (const float*)d_in[10];
    const float* wd2    = (const float*)d_in[11];
    const float* bd2    = (const float*)d_in[12];
    const float* wt     = (const float*)d_in[13];
    const float* bt     = (const float*)d_in[14];
    // d_in[15]=wz, d_in[16]=bz: dead code in reference (z never used)
    const float* wc     = (const float*)d_in[17];
    const float* bc     = (const float*)d_in[18];
    const float* w_prev = (const float*)d_in[19];
    const float* crisis = (const float*)d_in[20];

    char* ws = (char*)d_ws;
    __bf16* pw1  = (__bf16*)ws;                       // 144 frags * 1024 B = 147456
    __bf16* pw2e = (__bf16*)(ws + 147456);            // 416 frags * 1024 B = 425984
    float*  conc = (float*)(ws + 147456 + 425984);    // 240 floats
    float*  simb = conc + 240;                        // 48 floats

    pack_weights<<<70, 256, 0, stream>>>(we1, we2, pk, pw1, pw2e);
    conc_kernel<<<1, 1024, 0, stream>>>(pk, wd1, bd1, wd2, bd2, be2, conc, simb);
    actor_main<<<131072 / 32, 256, 0, stream>>>(
        state, fitness, be1, ln_g, ln_b, be2, pk, wt, bt, wc, bc,
        w_prev, crisis, pw1, pw2e, conc, simb, (float*)d_out);
}
